// EmergentCubeTrainer_42073499632191
// MI455X (gfx1250) — matrix operand order repaired and checked by reading, not compiled, not run on hardware
//
#include <hip/hip_runtime.h>
#include <math.h>

typedef __attribute__((ext_vector_type(16))) _Float16 v16h;
typedef __attribute__((ext_vector_type(2)))  _Float16 v2h;
typedef __attribute__((ext_vector_type(8)))  float    v8f;

#define NW 8  // waves per block (256 threads)

__device__ __forceinline__ v8f wmma_f16(v16h a, v16h b, v8f c) {
  return __builtin_amdgcn_wmma_f32_16x16x32_f16(false, a, false, b, (short)0, c, false, false);
}

// A-matrix 16x32 f16 layout: lane half 'hb' -> base = 8*hb.
// Slot i (0..7) holds K pair starting at: base + (i<4 ? 2i : 16+2(i-4))
__device__ __forceinline__ int a_slot_k(int i, int base) {
  return base + (i < 4 ? 2 * i : 16 + 2 * (i - 4));
}

// Load A fragment from row-major f32 (row r, 32 consecutive columns, row stride 'stride')
__device__ __forceinline__ v16h load_a_f32(const float* src, int stride, int r, int base) {
  v16h a;
#pragma unroll
  for (int i = 0; i < 8; ++i) {
    int k = a_slot_k(i, base);
    float2 f = *(const float2*)(src + r * stride + k);
    a[2 * i]     = (_Float16)f.x;
    a[2 * i + 1] = (_Float16)f.y;
  }
  return a;
}

// Last K-chunk of x: external_input (12 valid cols) + zero pad to 32
__device__ __forceinline__ v16h load_a_ext(const float* ext, int r, int base) {
  v16h a;
#pragma unroll
  for (int i = 0; i < 8; ++i) {
    int k = a_slot_k(i, base);
#pragma unroll
    for (int j = 0; j < 2; ++j) {
      int kk = k + j;
      float v = (kk < 12) ? ext[r * 12 + kk] : 0.0f;
      a[2 * i + j] = (_Float16)v;
    }
  }
  return a;
}

// Fragment = 32 lanes x 16 f16, per-lane contiguous (32B) -> 2x ds_load_b128
__device__ __forceinline__ v16h load_frag(const _Float16* frag, int lane) {
  return ((const v16h*)frag)[lane];
}

// Pre-swizzle one B fragment (32x16 f16 tile of a row-major f32 weight) into
// WMMA register layout. Exactly one dword slot per thread (tid 0..255).
__device__ __forceinline__ void fill_frag(_Float16* dst, const float* W, int ncols,
                                          int kvalid, int nvalid, int k0, int n0, int tid) {
  int lane = tid >> 3;   // 0..31
  int i    = tid & 7;    // 0..7
  int n  = (lane & 15) + n0;
  int kb = k0 + (lane >> 4) * 8 + (i < 4 ? 2 * i : 16 + 2 * (i - 4));
  float x0 = (kb     < kvalid && n < nvalid) ? W[kb * ncols + n]       : 0.0f;
  float x1 = (kb + 1 < kvalid && n < nvalid) ? W[(kb + 1) * ncols + n] : 0.0f;
  v2h p; p[0] = (_Float16)x0; p[1] = (_Float16)x1;
  *(v2h*)(dst + lane * 16 + 2 * i) = p;
}

// C-layout value (row m, col n) -> index in A-fragment layout buffer
__device__ __forceinline__ int afrag_idx(int m, int n) {
  int half = (n >> 3) & 1;                       // which 16-lane group holds col n
  int i = ((n & 7) >> 1) + ((n >= 16) ? 4 : 0);  // slot
  int j = n & 1;
  return (m + 16 * half) * 16 + 2 * i + j;
}

__global__ __launch_bounds__(256) void emergent_cube_kernel(
    const float* __restrict__ neighbor,  // [B,6,32]
    const float* __restrict__ own,       // [B,32]
    const float* __restrict__ ext,       // [B,12]
    const float* __restrict__ lctx,      // [B,32]
    const float* __restrict__ W1,        // [236,32]
    const float* __restrict__ b1,        // [32]
    const float* __restrict__ W2,        // [32,32]
    const float* __restrict__ b2,        // [32]
    const float* __restrict__ Wg1,       // [64,32]
    const float* __restrict__ bg1,       // [32]
    const float* __restrict__ Wg2,       // [32,6]
    const float* __restrict__ bg2,       // [6]
    const float* __restrict__ Wc,        // [32,32]
    const float* __restrict__ bc,        // [32]
    float* __restrict__ out)             // [B,32]
{
  // Fragment table (each 32 lanes x 16 f16 = 1KB):
  //  0..15 : W1  chunk c (0..7) x ntile t (0..1), idx = 2c+t
  // 16..17 : Wg1 top    (k0=0)  ntile 0/1
  // 18..19 : Wg1 bottom (k0=32) ntile 0/1
  // 20..21 : W2  ntile 0/1
  // 22     : Wg2 (padded N 6->16)
  // 23..24 : Wc  ntile 0/1
  __shared__ __align__(32) _Float16 fragW[25][512];
  __shared__ __align__(32) _Float16 act_frag[NW][512];  // activation staging, A-frag layout
  __shared__ float logit_lds[NW][16 * 6];
  __shared__ float w_lds[NW][6 * 16];
  __shared__ float spat_lds[NW][16 * 32];

  const int tid = threadIdx.x;

  // ---- pre-swizzle weights into WMMA fragment layout (one slot/thread/frag) ----
#pragma unroll
  for (int c = 0; c < 8; ++c) {
    fill_frag(fragW[2 * c],     W1, 32, 236, 32, c * 32, 0,  tid);
    fill_frag(fragW[2 * c + 1], W1, 32, 236, 32, c * 32, 16, tid);
  }
  fill_frag(fragW[16], Wg1, 32, 64, 32, 0,  0,  tid);
  fill_frag(fragW[17], Wg1, 32, 64, 32, 0,  16, tid);
  fill_frag(fragW[18], Wg1, 32, 64, 32, 32, 0,  tid);
  fill_frag(fragW[19], Wg1, 32, 64, 32, 32, 16, tid);
  fill_frag(fragW[20], W2, 32, 32, 32, 0, 0,  tid);
  fill_frag(fragW[21], W2, 32, 32, 32, 0, 16, tid);
  fill_frag(fragW[22], Wg2, 6, 32, 6, 0, 0, tid);
  fill_frag(fragW[23], Wc, 32, 32, 32, 0, 0,  tid);
  fill_frag(fragW[24], Wc, 32, 32, 32, 0, 16, tid);
  __syncthreads();

  const int lane = tid & 31;
  const int wid  = tid >> 5;
  const int r    = lane & 15;   // A row within tile / C-D column
  const int hb   = lane >> 4;
  const int base = hb * 8;

  const int row0 = (blockIdx.x * NW + wid) * 16;
  const float* nbrp = neighbor + (size_t)row0 * 192;
  const float* ownp = own      + (size_t)row0 * 32;
  const float* extp = ext      + (size_t)row0 * 12;
  const float* lcp  = lctx     + (size_t)row0 * 32;

  // ---- A fragments for x's K-chunks (neighbors reused later) ----
  v16h afr[8];
#pragma unroll
  for (int c = 0; c < 6; ++c) afr[c] = load_a_f32(nbrp + c * 32, 192, r, base);
  afr[6] = load_a_f32(ownp, 32, r, base);
  afr[7] = load_a_ext(extp, r, base);

  // ---- GEMM1: x @ W1 (K padded to 256, N=32) ----
  v8f acc1a = {}, acc1b = {};
#pragma unroll
  for (int c = 0; c < 8; ++c) {
    acc1a = wmma_f16(afr[c], load_frag(fragW[2 * c],     lane), acc1a);
    acc1b = wmma_f16(afr[c], load_frag(fragW[2 * c + 1], lane), acc1b);
  }

  // +b1, exact GELU, stage to LDS directly in A-fragment layout
#pragma unroll
  for (int nt = 0; nt < 2; ++nt) {
    v8f acc = nt ? acc1b : acc1a;
    int n = r + nt * 16;
    float bias = b1[n];
#pragma unroll
    for (int v = 0; v < 8; ++v) {
      int m = v + 8 * hb;
      float x = acc[v] + bias;
      float g = 0.5f * x * (1.0f + erff(x * 0.70710678118654752f));
      act_frag[wid][afrag_idx(m, n)] = (_Float16)g;
    }
  }
  __syncthreads();

  // ---- GEMM2: gelu(x@W1+b1) @ W2 ----
  v16h afrAct = load_frag(act_frag[wid], lane);
  v8f accB0 = {}, accB1 = {};
  accB0 = wmma_f16(afrAct, load_frag(fragW[20], lane), accB0);
  accB1 = wmma_f16(afrAct, load_frag(fragW[21], lane), accB1);

  // ---- gating: t = own @ Wg1[0:32] (shared across k) ----
  v8f tacc0 = {}, tacc1 = {};
  tacc0 = wmma_f16(afr[6], load_frag(fragW[16], lane), tacc0);
  tacc1 = wmma_f16(afr[6], load_frag(fragW[17], lane), tacc1);

  // hoist loop-invariant fragments into VGPRs
  v16h bWg1b0 = load_frag(fragW[18], lane);
  v16h bWg1b1 = load_frag(fragW[19], lane);
  v16h bWg2   = load_frag(fragW[22], lane);

  for (int k = 0; k < 6; ++k) {
    __syncthreads();  // act_frag[wid] free to reuse
    // h = relu(nbr_k @ Wg1[32:64] + t + bg1)
    v8f h0 = wmma_f16(afr[k], bWg1b0, tacc0);
    v8f h1 = wmma_f16(afr[k], bWg1b1, tacc1);
#pragma unroll
    for (int nt = 0; nt < 2; ++nt) {
      v8f hh = nt ? h1 : h0;
      int n = r + nt * 16;
      float bias = bg1[n];
#pragma unroll
      for (int v = 0; v < 8; ++v) {
        int m = v + 8 * hb;
        float x = hh[v] + bias;
        act_frag[wid][afrag_idx(m, n)] = (_Float16)(x > 0.0f ? x : 0.0f);
      }
    }
    __syncthreads();
    // logits = h @ Wg2 (cols 0..5 valid)
    v16h afrH = load_frag(act_frag[wid], lane);
    v8f lacc = {};
    lacc = wmma_f16(afrH, bWg2, lacc);
    if (r < 6) {
#pragma unroll
      for (int v = 0; v < 8; ++v) {
        int m = v + 8 * hb;
        logit_lds[wid][m * 6 + r] = lacc[v] + bg2[r];
      }
    }
    __syncthreads();
    // per-row softmax over 6 logits; keep diagonal element k
    if (lane < 16) {
      float l[6];
#pragma unroll
      for (int j = 0; j < 6; ++j) l[j] = logit_lds[wid][lane * 6 + j];
      float mx = l[0];
#pragma unroll
      for (int j = 1; j < 6; ++j) mx = fmaxf(mx, l[j]);
      float s = 0.0f, ek = 0.0f;
#pragma unroll
      for (int j = 0; j < 6; ++j) {
        float e = expf(l[j] - mx);
        s += e;
        if (j == k) ek = e;
      }
      w_lds[wid][k * 16 + lane] = ek / s;
    }
  }
  __syncthreads();

  // ---- spatial influence: sum_k w[k][row] * nbr_k (A-layout regs), transpose via LDS ----
  float spat[16];
#pragma unroll
  for (int i = 0; i < 16; ++i) spat[i] = 0.0f;
#pragma unroll
  for (int k = 0; k < 6; ++k) {
    float w = w_lds[wid][k * 16 + r];
    v16h a = afr[k];
#pragma unroll
    for (int i = 0; i < 16; ++i) spat[i] += w * (float)a[i];
  }
#pragma unroll
  for (int i = 0; i < 8; ++i) {
    int n = a_slot_k(i, base);
    float2 p; p.x = spat[2 * i]; p.y = spat[2 * i + 1];
    *(float2*)(&spat_lds[wid][r * 32 + n]) = p;
  }

  // ---- cross-layer: layer_context @ Wc ----
  v16h afrLC = load_a_f32(lcp, 32, r, base);
  v8f accC0 = {}, accC1 = {};
  accC0 = wmma_f16(afrLC, load_frag(fragW[23], lane), accC0);
  accC1 = wmma_f16(afrLC, load_frag(fragW[24], lane), accC1);
  __syncthreads();

  // ---- combine & store (C-layout) ----
#pragma unroll
  for (int nt = 0; nt < 2; ++nt) {
    int n = r + nt * 16;
    float b2v = b2[n];
    float bcv = bc[n];
#pragma unroll
    for (int v = 0; v < 8; ++v) {
      int m = v + 8 * hb;
      float basev = (nt ? accB1[v] : accB0[v]) + b2v;
      float sp    = spat_lds[wid][m * 32 + n] * (0.1f / 6.0f);
      float cl    = 0.05f * ((nt ? accC1[v] : accC0[v]) + bcv);
      out[(size_t)(row0 + m) * 32 + n] = basev + sp + cl;
    }
  }
}

extern "C" void kernel_launch(void* const* d_in, const int* in_sizes, int n_in,
                              void* d_out, int out_size, void* d_ws, size_t ws_size,
                              hipStream_t stream) {
  const float* neighbor = (const float*)d_in[0];
  const float* own      = (const float*)d_in[1];
  const float* ext      = (const float*)d_in[2];
  const float* lctx     = (const float*)d_in[3];
  const float* W1  = (const float*)d_in[4];
  const float* b1  = (const float*)d_in[5];
  const float* W2  = (const float*)d_in[6];
  const float* b2  = (const float*)d_in[7];
  const float* Wg1 = (const float*)d_in[8];
  const float* bg1 = (const float*)d_in[9];
  const float* Wg2 = (const float*)d_in[10];
  const float* bg2 = (const float*)d_in[11];
  const float* Wc  = (const float*)d_in[12];
  const float* bc  = (const float*)d_in[13];
  float* out = (float*)d_out;

  int B = in_sizes[1] / 32;          // own_state is [B,32]
  int blocks = B / (16 * NW);        // B = 262144 -> 2048 blocks, exact
  emergent_cube_kernel<<<blocks, 256, 0, stream>>>(
      neighbor, own, ext, lctx, W1, b1, W2, b2, Wg1, bg1, Wg2, bg2, Wc, bc, out);
}
